// GATEncoder_893353197859
// MI455X (gfx1250) — compile-verified, matrix operand order checked by reading, not run on hardware
//
#include <hip/hip_runtime.h>
#include <cstdint>
#include <cstddef>

#define N_NODES   50000
#define N_EDGES   1600000
#define EP        (N_EDGES + N_NODES)   // edges + self loops
#define IN_DIM    128
#define HID_DIM   128
#define OUT_DIM   64
#define HEADS     8
#define HPC       16                    // per-head channels, layer 1
#define NEG_SLOPE 0.2f
#define BN_EPS    1e-5f

typedef _Float16 v16h __attribute__((ext_vector_type(16)));
typedef float    v8f  __attribute__((ext_vector_type(8)));

// ---------------------------------------------------------------------------
// Fragment packing: convert fp32 matrices into f16 WMMA fragment order so the
// GEMM inner loop is pure {b128 load, b128 load, v_wmma}.
//
// Packed layout (v16h units): frag[(tile * (K/32) + kstep) * 32 + lane][j]
//   A: tile = M/16 tile, lane&15 = row-in-tile, lane>>4 selects K group
//   B: tile = N/16 tile, lane&15 = col-in-tile
//   half j -> K offset = kstep*32 + (lane>>4)*8 + (j<8 ? j : 8+j)   (ISA 7.12.2)
// ---------------------------------------------------------------------------
__device__ __forceinline__ int frag_koff(int lane, int j) {
    return (lane >> 4) * 8 + (j < 8 ? j : 8 + j);
}

template<int K>
__global__ void pack_a_kernel(const float* __restrict__ A, _Float16* __restrict__ pA,
                              int ntiles_m) {
    const int gid = blockIdx.x * blockDim.x + threadIdx.x;
    const int total = ntiles_m * (K / 32) * 32 * 16;
    if (gid >= total) return;
    const int j    = gid & 15;
    const int lane = (gid >> 4) & 31;
    const int rest = gid >> 9;
    const int ks   = rest % (K / 32);
    const int t    = rest / (K / 32);
    const int row  = t * 16 + (lane & 15);
    const int kk   = ks * 32 + frag_koff(lane, j);
    pA[gid] = (_Float16)A[(size_t)row * K + kk];
}

template<int K, int NCOLS>
__global__ void pack_b_kernel(const float* __restrict__ B, _Float16* __restrict__ pB) {
    const int gid = blockIdx.x * blockDim.x + threadIdx.x;
    const int total = (NCOLS / 16) * (K / 32) * 32 * 16;
    if (gid >= total) return;
    const int j    = gid & 15;
    const int lane = (gid >> 4) & 31;
    const int rest = gid >> 9;
    const int ks   = rest % (K / 32);
    const int nt   = rest / (K / 32);
    const int col  = nt * 16 + (lane & 15);
    const int kk   = ks * 32 + frag_koff(lane, j);
    pB[gid] = (_Float16)B[(size_t)kk * NCOLS + col];
}

// ---------------------------------------------------------------------------
// WMMA GEMM on pre-packed fragments. One wave per 16x16 C tile.
// blockDim.x = 32 * (NCOLS/16); blockIdx.x = M tile.
// ---------------------------------------------------------------------------
template<int K, int NCOLS>
__global__ void gemm_wmma_kernel(const _Float16* __restrict__ pA,
                                 const _Float16* __restrict__ pB,
                                 float* __restrict__ C) {
    constexpr int KS = K / 32;
    const int lane = threadIdx.x & 31;
    const int wave = threadIdx.x >> 5;            // = N tile
    const int mt   = blockIdx.x;

    const v16h* __restrict__ Af = reinterpret_cast<const v16h*>(pA) + (size_t)mt   * KS * 32 + lane;
    const v16h* __restrict__ Bf = reinterpret_cast<const v16h*>(pB) + (size_t)wave * KS * 32 + lane;

    v8f acc = {};
    #pragma unroll
    for (int ks = 0; ks < KS; ++ks) {
        const v16h af = Af[ks * 32];
        const v16h bf = Bf[ks * 32];
        acc = __builtin_amdgcn_wmma_f32_16x16x32_f16(
            false, af, false, bf, (short)0, acc, false, false);
    }

    const int grp = lane >> 4;
    const int col = wave * 16 + (lane & 15);
    #pragma unroll
    for (int r = 0; r < 8; ++r) {
        const int row = mt * 16 + r + 8 * grp;    // C layout: VGPR r -> M=r / M=r+8
        C[(size_t)row * NCOLS + col] = acc[r];
    }
}

// ---------------------------------------------------------------------------
// Per-node attention logit halves
// ---------------------------------------------------------------------------
__global__ void alpha1_kernel(const float* __restrict__ h,
                              const float* __restrict__ a_src,
                              const float* __restrict__ a_dst,
                              float* __restrict__ as, float* __restrict__ ad) {
    const int idx = blockIdx.x * blockDim.x + threadIdx.x;
    if (idx >= N_NODES * HEADS) return;
    const int n = idx / HEADS, hh = idx % HEADS;
    const float* hp = h + (size_t)n * HID_DIM + hh * HPC;
    float s = 0.f, d = 0.f;
    #pragma unroll
    for (int c = 0; c < HPC; ++c) {
        const float v = hp[c];
        s += v * a_src[hh * HPC + c];
        d += v * a_dst[hh * HPC + c];
    }
    as[idx] = s; ad[idx] = d;
}

__global__ void alpha2_kernel(const float* __restrict__ h2,
                              const float* __restrict__ a_src,
                              const float* __restrict__ a_dst,
                              float* __restrict__ as, float* __restrict__ ad) {
    const int n = blockIdx.x * blockDim.x + threadIdx.x;
    if (n >= N_NODES) return;
    const float* hp = h2 + (size_t)n * OUT_DIM;
    float s = 0.f, d = 0.f;
    #pragma unroll 8
    for (int c = 0; c < OUT_DIM; ++c) {
        const float v = hp[c];
        s += v * a_src[c];
        d += v * a_dst[c];
    }
    as[n] = s; ad[n] = d;
}

// ---------------------------------------------------------------------------
// Edge helpers
// ---------------------------------------------------------------------------
__device__ __forceinline__ void get_edge(const int* __restrict__ ei, int e,
                                         int& s, int& d) {
    if (e < N_EDGES) { s = ei[e]; d = ei[N_EDGES + e]; }
    else             { s = d = e - N_EDGES; }          // appended self loop
}

__device__ __forceinline__ float leaky(float x) {
    return x > 0.f ? x : NEG_SLOPE * x;
}

// float atomic max via signed-max / unsigned-min ordering trick (non-NaN)
__device__ __forceinline__ void atomicMaxF(float* addr, float val) {
    if (val >= 0.f) atomicMax((int*)addr, __float_as_int(val));
    else            atomicMin((unsigned int*)addr, __float_as_uint(val));
}

template<int H>
__global__ void edge_max_kernel(const int* __restrict__ ei,
                                const float* __restrict__ as,
                                const float* __restrict__ ad,
                                float* __restrict__ m) {
    const int idx = blockIdx.x * blockDim.x + threadIdx.x;
    if (idx >= EP * H) return;
    const int e = idx / H, hh = idx % H;
    int s, d; get_edge(ei, e, s, d);
    const float x = leaky(as[s * H + hh] + ad[d * H + hh]);
    atomicMaxF(&m[d * H + hh], x);
}

template<int H>
__global__ void edge_sum_kernel(const int* __restrict__ ei,
                                const float* __restrict__ as,
                                const float* __restrict__ ad,
                                const float* __restrict__ m,
                                float* __restrict__ den) {
    const int idx = blockIdx.x * blockDim.x + threadIdx.x;
    if (idx >= EP * H) return;
    const int e = idx / H, hh = idx % H;
    int s, d; get_edge(ei, e, s, d);
    const float x = leaky(as[s * H + hh] + ad[d * H + hh]);
    atomicAdd(&den[d * H + hh], __expf(x - m[d * H + hh]));
}

// One wave32 per edge; 4 channels per lane (128 channels).
__global__ void scatter1_kernel(const int* __restrict__ ei,
                                const float* __restrict__ h,
                                const float* __restrict__ as,
                                const float* __restrict__ ad,
                                const float* __restrict__ m,
                                const float* __restrict__ den,
                                float* __restrict__ out) {
    const int wid  = (blockIdx.x * blockDim.x + threadIdx.x) >> 5;
    const int lane = threadIdx.x & 31;
    if (wid >= EP) return;
    int s, d; get_edge(ei, wid, s, d);
    const int ch = lane * 4;
    const int hh = ch >> 4;   // head = channel / 16
    const float x  = leaky(as[s * HEADS + hh] + ad[d * HEADS + hh]);
    const float ee = __expf(x - m[d * HEADS + hh]);
    const float al = ee / (den[d * HEADS + hh] + 1e-16f);
    const float4 hv = *reinterpret_cast<const float4*>(h + (size_t)s * HID_DIM + ch);
    float* op = out + (size_t)d * HID_DIM + ch;
    atomicAdd(op + 0, hv.x * al);
    atomicAdd(op + 1, hv.y * al);
    atomicAdd(op + 2, hv.z * al);
    atomicAdd(op + 3, hv.w * al);
}

// One wave32 per edge; 2 channels per lane (64 channels), single head.
__global__ void scatter2_kernel(const int* __restrict__ ei,
                                const float* __restrict__ h,
                                const float* __restrict__ as,
                                const float* __restrict__ ad,
                                const float* __restrict__ m,
                                const float* __restrict__ den,
                                float* __restrict__ out) {
    const int wid  = (blockIdx.x * blockDim.x + threadIdx.x) >> 5;
    const int lane = threadIdx.x & 31;
    if (wid >= EP) return;
    int s, d; get_edge(ei, wid, s, d);
    const float x  = leaky(as[s] + ad[d]);
    const float ee = __expf(x - m[d]);
    const float al = ee / (den[d] + 1e-16f);
    const int ch = lane * 2;
    const float2 hv = *reinterpret_cast<const float2*>(h + (size_t)s * OUT_DIM + ch);
    float* op = out + (size_t)d * OUT_DIM + ch;
    atomicAdd(op + 0, hv.x * al);
    atomicAdd(op + 1, hv.y * al);
}

// ---------------------------------------------------------------------------
// BatchNorm (+ bias fold) + ELU
// ---------------------------------------------------------------------------
__global__ void bn_stats_kernel(const float* __restrict__ out1,
                                const float* __restrict__ b1,
                                float* __restrict__ sum, float* __restrict__ sumsq) {
    const int c  = threadIdx.x;                   // 128 threads = channels
    const int r0 = blockIdx.x * 64;
    const int r1 = (r0 + 64 < N_NODES) ? r0 + 64 : N_NODES;
    const float bias = b1[c];
    float s = 0.f, ss = 0.f;
    for (int r = r0; r < r1; ++r) {
        const float v = out1[(size_t)r * HID_DIM + c] + bias;
        s += v; ss += v * v;
    }
    atomicAdd(&sum[c], s);
    atomicAdd(&sumsq[c], ss);
}

__global__ void bn_finalize_kernel(const float* __restrict__ sum,
                                   const float* __restrict__ sumsq,
                                   const float* __restrict__ gamma,
                                   const float* __restrict__ beta,
                                   float* __restrict__ scale, float* __restrict__ shift) {
    const int c = threadIdx.x;                    // single block, 128 threads
    const float invN = 1.0f / (float)N_NODES;
    const float mu   = sum[c] * invN;
    const float var  = sumsq[c] * invN - mu * mu;
    const float sc   = gamma[c] * rsqrtf(var + BN_EPS);
    scale[c] = sc;
    shift[c] = beta[c] - mu * sc;
}

__global__ void bn_apply_elu_kernel(const float* __restrict__ out1,
                                    const float* __restrict__ b1,
                                    const float* __restrict__ scale,
                                    const float* __restrict__ shift,
                                    float* __restrict__ hbn) {
    const int idx = blockIdx.x * blockDim.x + threadIdx.x;
    if (idx >= N_NODES * HID_DIM) return;
    const int c = idx & (HID_DIM - 1);
    float v = out1[idx] + b1[c];
    v = v * scale[c] + shift[c];
    hbn[idx] = v > 0.f ? v : __expf(v) - 1.f;     // ELU(alpha=1)
}

// ---------------------------------------------------------------------------
// Init helpers
// ---------------------------------------------------------------------------
__global__ void fill_kernel(float* __restrict__ p, int n, float v) {
    const int i = blockIdx.x * blockDim.x + threadIdx.x;
    if (i < n) p[i] = v;
}

__global__ void out_bias_init_kernel(float* __restrict__ out, const float* __restrict__ b2) {
    const int i = blockIdx.x * blockDim.x + threadIdx.x;
    if (i < N_NODES * OUT_DIM) out[i] = b2[i & (OUT_DIM - 1)];
}

// ---------------------------------------------------------------------------
// Launch
// ---------------------------------------------------------------------------
static inline int cdiv(long long a, long long b) { return (int)((a + b - 1) / b); }

extern "C" void kernel_launch(void* const* d_in, const int* in_sizes, int n_in,
                              void* d_out, int out_size, void* d_ws, size_t ws_size,
                              hipStream_t stream) {
    const float* x       = (const float*)d_in[0];
    const int*   ei      = (const int*)  d_in[1];   // [2, N_EDGES]
    const float* W1      = (const float*)d_in[2];
    const float* a1_src  = (const float*)d_in[3];
    const float* a1_dst  = (const float*)d_in[4];
    const float* b1      = (const float*)d_in[5];
    const float* bn_g    = (const float*)d_in[6];
    const float* bn_b    = (const float*)d_in[7];
    const float* W2      = (const float*)d_in[8];
    const float* a2_src  = (const float*)d_in[9];
    const float* a2_dst  = (const float*)d_in[10];
    const float* b2      = (const float*)d_in[11];
    float* out = (float*)d_out;

    const int MT = N_NODES / 16;                    // 3125 M tiles (exact)

    // workspace carve-up (floats; every region a multiple of 32B for v16h loads)
    float* ws = (float*)d_ws;
    size_t off = 0;
    float* h1    = ws + off; off += (size_t)N_NODES * HID_DIM;   // X@W1
    float* out1  = ws + off; off += (size_t)N_NODES * HID_DIM;   // layer-1 aggregate
    float* hbn   = ws + off; off += (size_t)N_NODES * HID_DIM;   // BN+ELU output
    float* h2    = ws + off; off += (size_t)N_NODES * OUT_DIM;   // hbn@W2
    _Float16* pA1 = (_Float16*)(ws + off); off += (size_t)N_NODES * IN_DIM  / 2;  // packed x
    _Float16* pA2 = (_Float16*)(ws + off); off += (size_t)N_NODES * HID_DIM / 2;  // packed hbn
    _Float16* pB1 = (_Float16*)(ws + off); off += (size_t)IN_DIM  * HID_DIM / 2;  // packed W1
    _Float16* pB2 = (_Float16*)(ws + off); off += (size_t)HID_DIM * OUT_DIM / 2;  // packed W2
    float* as1   = ws + off; off += (size_t)N_NODES * HEADS;
    float* ad1   = ws + off; off += (size_t)N_NODES * HEADS;
    float* m1    = ws + off; off += (size_t)N_NODES * HEADS;
    float* den1  = ws + off; off += (size_t)N_NODES * HEADS;
    float* as2   = ws + off; off += (size_t)N_NODES;
    float* ad2   = ws + off; off += (size_t)N_NODES;
    float* m2    = ws + off; off += (size_t)N_NODES;
    float* den2  = ws + off; off += (size_t)N_NODES;
    float* sum   = ws + off; off += HID_DIM;
    float* sumsq = ws + off; off += HID_DIM;
    float* scale = ws + off; off += HID_DIM;
    float* shift = ws + off; off += HID_DIM;

    const int NH  = N_NODES * HEADS;
    const int NE1 = EP * HEADS;

    // ---- Layer 1 ----
    pack_a_kernel<IN_DIM><<<cdiv((long long)N_NODES * IN_DIM, 256), 256, 0, stream>>>(x, pA1, MT);
    pack_b_kernel<IN_DIM, HID_DIM><<<cdiv(IN_DIM * HID_DIM, 256), 256, 0, stream>>>(W1, pB1);
    gemm_wmma_kernel<IN_DIM, HID_DIM><<<MT, 256, 0, stream>>>(pA1, pB1, h1);
    alpha1_kernel<<<cdiv(NH, 256), 256, 0, stream>>>(h1, a1_src, a1_dst, as1, ad1);

    fill_kernel<<<cdiv(NH, 256), 256, 0, stream>>>(m1, NH, -INFINITY);
    fill_kernel<<<cdiv(NH, 256), 256, 0, stream>>>(den1, NH, 0.f);
    fill_kernel<<<cdiv((long long)N_NODES * HID_DIM, 256), 256, 0, stream>>>(out1, N_NODES * HID_DIM, 0.f);
    fill_kernel<<<1, 256, 0, stream>>>(sum, HID_DIM, 0.f);
    fill_kernel<<<1, 256, 0, stream>>>(sumsq, HID_DIM, 0.f);

    edge_max_kernel<HEADS><<<cdiv(NE1, 256), 256, 0, stream>>>(ei, as1, ad1, m1);
    edge_sum_kernel<HEADS><<<cdiv(NE1, 256), 256, 0, stream>>>(ei, as1, ad1, m1, den1);
    scatter1_kernel<<<cdiv((long long)EP * 32, 256), 256, 0, stream>>>(ei, h1, as1, ad1, m1, den1, out1);

    bn_stats_kernel<<<cdiv(N_NODES, 64), HID_DIM, 0, stream>>>(out1, b1, sum, sumsq);
    bn_finalize_kernel<<<1, HID_DIM, 0, stream>>>(sum, sumsq, bn_g, bn_b, scale, shift);
    bn_apply_elu_kernel<<<cdiv((long long)N_NODES * HID_DIM, 256), 256, 0, stream>>>(out1, b1, scale, shift, hbn);

    // ---- Layer 2 ----
    pack_a_kernel<HID_DIM><<<cdiv((long long)N_NODES * HID_DIM, 256), 256, 0, stream>>>(hbn, pA2, MT);
    pack_b_kernel<HID_DIM, OUT_DIM><<<cdiv(HID_DIM * OUT_DIM, 256), 256, 0, stream>>>(W2, pB2);
    gemm_wmma_kernel<HID_DIM, OUT_DIM><<<MT, 128, 0, stream>>>(pA2, pB2, h2);
    alpha2_kernel<<<cdiv(N_NODES, 256), 256, 0, stream>>>(h2, a2_src, a2_dst, as2, ad2);

    fill_kernel<<<cdiv(N_NODES, 256), 256, 0, stream>>>(m2, N_NODES, -INFINITY);
    fill_kernel<<<cdiv(N_NODES, 256), 256, 0, stream>>>(den2, N_NODES, 0.f);
    out_bias_init_kernel<<<cdiv((long long)N_NODES * OUT_DIM, 256), 256, 0, stream>>>(out, b2);

    edge_max_kernel<1><<<cdiv(EP, 256), 256, 0, stream>>>(ei, as2, ad2, m2);
    edge_sum_kernel<1><<<cdiv(EP, 256), 256, 0, stream>>>(ei, as2, ad2, m2, den2);
    scatter2_kernel<<<cdiv((long long)EP * 32, 256), 256, 0, stream>>>(ei, h2, as2, ad2, m2, den2, out);
}
